// DRNNComponent_8435315769655
// MI455X (gfx1250) — matrix operand order repaired and checked by reading, not compiled, not run on hardware
//
#include <hip/hip_runtime.h>

typedef __attribute__((ext_vector_type(16))) __bf16 v16bf;
typedef __attribute__((ext_vector_type(8)))  float  v8f;

typedef unsigned int   u32;
typedef unsigned short u16;

static constexpr int Bb = 128;   // batch
static constexpr int Tt = 1024;  // time
static constexpr int Ff = 128;   // input features
static constexpr int Hh = 256;   // hidden
static constexpr int G4 = 1024;  // 4*H gates

struct Frag32 { uint4 lo; uint4 hi; };   // 32 bytes = one 16xbf16-per-lane operand slice

static __device__ __forceinline__ u16 f2bf(float f) {
  u32 u = __builtin_bit_cast(u32, f);
  u32 r = (u + 0x7FFFu + ((u >> 16) & 1u)) >> 16;   // round-to-nearest-even
  return (u16)r;
}
static __device__ __forceinline__ u32 pack2(float a, float b) {
  return (u32)f2bf(a) | ((u32)f2bf(b) << 16);
}
static __device__ __forceinline__ float sigm(float x) {
  return 1.0f / (1.0f + __expf(-x));
}

// ---------------- prep kernels ----------------

__global__ void zero_u32(u32* p, int n) {
  int i = blockIdx.x * blockDim.x + threadIdx.x;
  if (i < n) p[i] = 0u;
}

__global__ void bias_sum(const float* __restrict__ a, const float* __restrict__ b,
                         float* __restrict__ o) {
  int i = blockIdx.x * blockDim.x + threadIdx.x;
  if (i < G4) o[i] = a[i] + b[i];
}

// Pack combined [Wih | Whh] (row-major f32, rows = 4H gate outputs) into the
// CDNA5 WMMA bf16 B-operand shape, using a contiguous-K order matched by A, fragment-contiguous:
//   frag(nt, kt): 32 lanes x 8 dwords, lane = nt*16 + (lane%16) output column,
//   K group = kt*32 + (lane/16)*16 + 2d  (16 contiguous K per lane).
__global__ void pack_weights(const float* __restrict__ wih,
                             const float* __restrict__ whh,
                             u32* __restrict__ dst, int kin, int ktiles) {
  int idx = blockIdx.x * blockDim.x + threadIdx.x;
  int total = 64 * ktiles * 256;
  if (idx >= total) return;
  int frag = idx >> 8;
  int s    = idx & 255;
  int lam  = s >> 3;
  int d    = s & 7;
  int nt   = frag / ktiles;
  int kt   = frag - nt * ktiles;
  int n    = nt * 16 + (lam & 15);
  int k0   = kt * 32 + (lam >> 4) * 16 + 2 * d;
  float f0 = (k0 < kin) ? wih[n * kin + k0] : whh[n * Hh + (k0 - kin)];
  float f1 = (k0 + 1 < kin) ? wih[n * kin + k0 + 1] : whh[n * Hh + (k0 + 1 - kin)];
  dst[idx] = pack2(f0, f1);
}

// ---------------- per-step LSTM layer kernel ----------------
// grid = 128 blocks (bi = batch tile 0..7, hj = hidden tile 0..15), 128 threads
// wave w (0..3) computes gate w's 16x16 tile: D = A[16xKL] x B[KLx16] via WMMA.

template <int KL, bool IS_L0, bool WRITE_Y>
__global__ __launch_bounds__(128) void lstm_step_kernel(
    const float* __restrict__ x,       // [B,T,F] f32 (layer 0 only)
    const u32*  __restrict__ aprev32,  // prev-layer h, bf16 pairs (layers 1/2)
    const u32*  __restrict__ hold32,   // own h_old, bf16 pairs
    u16*        __restrict__ hnew,     // own h_new, bf16
    float*      __restrict__ cst,      // own c state, f32
    const u32*  __restrict__ wpack,    // packed bf16 weights
    const float* __restrict__ bsum,    // fused bias [4H]
    float*      __restrict__ y,        // [B,T,H] f32 (layer 2 only)
    int t) {
  constexpr int KIN = KL - Hh;         // 128 (L0) or 256 (L1/L2)
  constexpr int SP  = KL + 8;          // padded LDS row stride in bf16 elems
  constexpr int KT  = KL / 32;         // WMMA K steps (12 or 16)

  __shared__ __align__(16) u16 A_s[16 * SP];
  __shared__ float G_s[4 * 256];

  const int tid = threadIdx.x;
  const int bi  = blockIdx.x & 7;
  const int hj  = blockIdx.x >> 3;

  // ---- stage A tile (16 batch rows x KL cols) into LDS as packed bf16 ----
  u32* As32 = (u32*)A_s;
  for (int i = tid; i < 16 * (KL / 2); i += 128) {
    int row = i / (KL / 2);
    int pr  = i - row * (KL / 2);
    int col = pr * 2;
    int rg  = bi * 16 + row;
    u32 v;
    if (col < KIN) {
      if (IS_L0) {
        const float2 xv = *(const float2*)(x + ((size_t)rg * Tt + t) * Ff + col);
        v = pack2(xv.x, xv.y);
      } else {
        v = aprev32[rg * (Hh / 2) + pr];
      }
    } else {
      v = hold32[rg * (Hh / 2) + (pr - KIN / 2)];
    }
    As32[row * (SP / 2) + pr] = v;
  }
  __syncthreads();

  // ---- WMMA accumulation: wave = gate, tile = (bi, hj) ----
  const int wave = tid >> 5;
  const int lane = tid & 31;
  const int m    = lane & 15;
  const int half = lane >> 4;
  const int nt   = wave * 16 + hj;  // gate-output tile index (0..63)

  const u32* wf   = wpack + (size_t)nt * KT * 256 + lane * 8;
  const u32* arow = As32 + m * (SP / 2);

  v8f acc = {};
#pragma unroll
  for (int kt = 0; kt < KT; ++kt) {
    // A fragment: lane<16 -> K {0..15}, lane>=16 -> K {16..31} (matches packed B)
    const uint4* ap = (const uint4*)(arow + kt * 16 + half * 8);
    Frag32 af;
    af.lo = ap[0];
    af.hi = ap[1];
    // B fragment: prepacked, two coalesced b128 loads per lane
    const uint4* bp = (const uint4*)(wf + kt * 256);
    Frag32 bf;
    bf.lo = bp[0];
    bf.hi = bp[1];
    acc = __builtin_amdgcn_wmma_f32_16x16x32_bf16(
        false, __builtin_bit_cast(v16bf, af),
        false, __builtin_bit_cast(v16bf, bf),
        (short)0, acc, false, false);
  }

  // ---- exchange gate tiles through LDS (C/D layout: lane%16 = N, vgpr r -> M) ----
  float* gw = G_s + wave * 256;
  const int nn = lane & 15;
#pragma unroll
  for (int r = 0; r < 8; ++r) gw[(r + half * 8) * 16 + nn] = acc[r];
  __syncthreads();

  // ---- pointwise LSTM update for this (bi, hj) 16x16 state tile ----
  for (int e = tid; e < 256; e += 128) {
    int mm  = e >> 4;
    int n2  = e & 15;
    int row = bi * 16 + mm;
    int col = hj * 16 + n2;
    float gi = G_s[e]       + bsum[col];
    float gf = G_s[256 + e] + bsum[Hh + col];
    float gg = G_s[512 + e] + bsum[2 * Hh + col];
    float go = G_s[768 + e] + bsum[3 * Hh + col];
    int ci   = (row << 8) + col;
    float c0 = cst[ci];
    float cn = sigm(gf) * c0 + sigm(gi) * tanhf(gg);
    float hn = sigm(go) * tanhf(cn);
    cst[ci]  = cn;
    hnew[ci] = f2bf(hn);
    if (WRITE_Y) {
      // layer 2 only updates every 4 steps -> y[t..t+3] all equal new h2
      float* yp = y + ((size_t)row * Tt + t) * Hh + col;
      yp[0]      = hn;
      yp[Hh]     = hn;
      yp[2 * Hh] = hn;
      yp[3 * Hh] = hn;
    }
  }
}

// ---------------- host ----------------

extern "C" void kernel_launch(void* const* d_in, const int* in_sizes, int n_in,
                              void* d_out, int out_size, void* d_ws, size_t ws_size,
                              hipStream_t stream) {
  (void)in_sizes; (void)n_in; (void)out_size; (void)ws_size;

  const float* x = (const float*)d_in[0];
  const float* Wih[3] = {(const float*)d_in[1], (const float*)d_in[5], (const float*)d_in[9]};
  const float* Whh[3] = {(const float*)d_in[2], (const float*)d_in[6], (const float*)d_in[10]};
  const float* bih[3] = {(const float*)d_in[3], (const float*)d_in[7], (const float*)d_in[11]};
  const float* bhh[3] = {(const float*)d_in[4], (const float*)d_in[8], (const float*)d_in[12]};
  float* y = (float*)d_out;

  // workspace layout (≈3.6 MB total)
  char*  ws  = (char*)d_ws;
  u16*   HBF = (u16*)ws;               // [3 layers][2 buffers][B][H] bf16 = 393216 B
  float* C   = (float*)(ws + 393216);  // [3][B][H] f32                 = 393216 B
  float* BS  = (float*)(ws + 786432);  // [3][4H] fused bias            =  12288 B
  u32*   WP[3];
  WP[0] = (u32*)(ws + 798720);         // layer0: 64*12*256 dwords
  WP[1] = WP[0] + 64 * 12 * 256;       // layer1: 64*16*256 dwords
  WP[2] = WP[1] + 64 * 16 * 256;       // layer2: 64*16*256 dwords

  auto hbuf = [&](int l, int p) { return HBF + (size_t)(l * 2 + p) * Bb * Hh; };

  // zero h (both buffers) + c state (contiguous region)
  {
    int nz = (393216 + 393216) / 4;
    zero_u32<<<(nz + 255) / 256, 256, 0, stream>>>((u32*)ws, nz);
  }
  for (int l = 0; l < 3; ++l)
    bias_sum<<<4, 256, 0, stream>>>(bih[l], bhh[l], BS + l * G4);
  pack_weights<<<768, 256, 0, stream>>>(Wih[0], Whh[0], WP[0], 128, 12);
  pack_weights<<<1024, 256, 0, stream>>>(Wih[1], Whh[1], WP[1], 256, 16);
  pack_weights<<<1024, 256, 0, stream>>>(Wih[2], Whh[2], WP[2], 256, 16);

  // sequential dilated recurrence; double-buffer h per layer
  int cur[3] = {0, 0, 0};
  for (int t = 0; t < Tt; ++t) {
    lstm_step_kernel<384, true, false><<<128, 128, 0, stream>>>(
        x, nullptr, (const u32*)hbuf(0, cur[0]), hbuf(0, cur[0] ^ 1),
        C, WP[0], BS, nullptr, t);
    cur[0] ^= 1;
    if ((t & 1) == 0) {
      lstm_step_kernel<512, false, false><<<128, 128, 0, stream>>>(
          nullptr, (const u32*)hbuf(0, cur[0]), (const u32*)hbuf(1, cur[1]),
          hbuf(1, cur[1] ^ 1), C + Bb * Hh, WP[1], BS + G4, nullptr, t);
      cur[1] ^= 1;
    }
    if ((t & 3) == 0) {
      lstm_step_kernel<512, false, true><<<128, 128, 0, stream>>>(
          nullptr, (const u32*)hbuf(1, cur[1]), (const u32*)hbuf(2, cur[2]),
          hbuf(2, cur[2] ^ 1), C + 2 * Bb * Hh, WP[2], BS + 2 * G4, y, t);
      cur[2] ^= 1;
    }
  }
}
